// HyperbolicTransformer_7576322310248
// MI455X (gfx1250) — compile-verified, matrix operand order checked by reading, not run on hardware
//
#include <hip/hip_runtime.h>
#include <cstdint>
#include <cstddef>

// ---------------------------------------------------------------------------
// HyperbolicTransformer forward for MI455X (gfx1250, wave32, WMMA).
// GEMMs (99% of FLOPs): bf16 WMMA 16x16x32, f32 accumulate, LDS double-buffer
// fed by GLOBAL_LOAD_ASYNC_TO_LDS_B128 (ASYNCcnt-tracked async copies).
// Hyperbolic geometry / softmax / LN / pooling stay in f32 VALU.
// ---------------------------------------------------------------------------

#define EPSV 1e-6f
#define MAXN (1.0f - 1e-5f)

typedef __attribute__((ext_vector_type(16))) __bf16 v16bf;
typedef __attribute__((ext_vector_type(8)))  __bf16 v8bf;
typedef __attribute__((ext_vector_type(8)))  float  v8f;

static constexpr int Bc  = 64;
static constexpr int Lc  = 100;
static constexpr int Dc  = 1024;
static constexpr int Cc  = 1000;
static constexpr int NLc = 2;
static constexpr int DFc = 4096;
static constexpr int Hc  = 16;
static constexpr int Mc  = Bc * Lc;   // 6400 rows

// ---------------------------------------------------------------- reductions
__device__ __forceinline__ float wred_sum(float v) {
#pragma unroll
  for (int o = 16; o > 0; o >>= 1) v += __shfl_xor(v, o, 32);
  return v;
}
__device__ __forceinline__ float wred_max(float v) {
#pragma unroll
  for (int o = 16; o > 0; o >>= 1) v = fmaxf(v, __shfl_xor(v, o, 32));
  return v;
}
// 256-thread block sum (8 wave32s)
__device__ __forceinline__ float block_sum_256(float v, float* sbuf) {
  int lane = threadIdx.x & 31, w = threadIdx.x >> 5;
  v = wred_sum(v);
  if (lane == 0) sbuf[w] = v;
  __syncthreads();
  float r = (threadIdx.x < 8) ? sbuf[threadIdx.x] : 0.0f;
  if (w == 0) {
#pragma unroll
    for (int o = 4; o > 0; o >>= 1) r += __shfl_xor(r, o, 32);
    if (lane == 0) sbuf[0] = r;
  }
  __syncthreads();
  float out = sbuf[0];
  __syncthreads();
  return out;
}

// ------------------------------------------------------------- convert f32->bf16
__global__ __launch_bounds__(256) void cvt_bf16_kernel(const float* __restrict__ in,
                                                       __bf16* __restrict__ out, int n) {
  int i = blockIdx.x * 256 + threadIdx.x;
  if (i < n) out[i] = (__bf16)in[i];
}

// ------------------------------------------------------------- GELU (exact) + cvt
__global__ __launch_bounds__(256) void gelu_cvt_kernel(const float* __restrict__ in,
                                                       __bf16* __restrict__ out, int n) {
  int i = blockIdx.x * 256 + threadIdx.x;
  if (i < n) {
    float x = in[i];
    float g = 0.5f * x * (1.0f + erff(x * 0.70710678118654752f));
    out[i] = (__bf16)g;
  }
}

// ------------------------------------------------------------- embedding + exp map
__global__ __launch_bounds__(256) void embed_kernel(const int* __restrict__ x,
                                                    const float* __restrict__ emb,
                                                    const float* __restrict__ pos,
                                                    float* __restrict__ h,
                                                    __bf16* __restrict__ hbf) {
  __shared__ float red[32];
  int row = blockIdx.x;                 // 0..Mc-1
  int l   = row % Lc;
  int tid = threadIdx.x;
  int tok = x[row];
  const float* er = emb + (size_t)tok * Dc;
  const float* pr = pos + (size_t)l * Dc;

  float e[4], v[4];
#pragma unroll
  for (int j = 0; j < 4; j++) {
    int c = tid + j * 256;
    e[j] = er[c];
    v[j] = pr[c] * 0.1f;
  }
  float s = 0.f;
#pragma unroll
  for (int j = 0; j < 4; j++) s += e[j] * e[j];
  s = block_sum_256(s, red);
  float nrm = sqrtf(s);
  if (nrm >= MAXN) {
    float sc = MAXN / fmaxf(nrm, EPSV);
#pragma unroll
    for (int j = 0; j < 4; j++) e[j] *= sc;
  }
  float x2p = 0.f, vnp = 0.f, evp = 0.f;
#pragma unroll
  for (int j = 0; j < 4; j++) { x2p += e[j]*e[j]; vnp += v[j]*v[j]; evp += e[j]*v[j]; }
  float x2  = block_sum_256(x2p, red);
  float vn2 = block_sum_256(vnp, red);
  float ev  = block_sum_256(evp, red);

  float lam = 2.0f / fmaxf(1.0f - x2, EPSV);
  float vn  = fmaxf(sqrtf(vn2), EPSV);
  float t   = tanhf(lam * vn * 0.5f);
  float tv  = t / vn;
  float xy  = tv * ev;
  float y2  = tv * tv * vn2;
  float den = fmaxf(1.0f + 2.0f * xy + x2 * y2, EPSV);
  float c1  = 1.0f + 2.0f * xy + y2;
  float c2  = 1.0f - x2;

  float hv[4]; float hn = 0.f;
#pragma unroll
  for (int j = 0; j < 4; j++) {
    hv[j] = (c1 * e[j] + c2 * tv * v[j]) / den;
    hn += hv[j] * hv[j];
  }
  hn = block_sum_256(hn, red);
  float nn  = sqrtf(hn);
  float sc2 = (nn < MAXN) ? 1.0f : MAXN / fmaxf(nn, EPSV);
#pragma unroll
  for (int j = 0; j < 4; j++) {
    int c = tid + j * 256;
    float y = hv[j] * sc2;
    h[(size_t)row * Dc + c]   = y;
    hbf[(size_t)row * Dc + c] = (__bf16)y;
  }
}

// ------------------------------------------------------------- WMMA GEMM
// C[M,N] = A[M,K](bf16) @ W[N,K]^T(bf16) + bias[N]; K % 32 == 0.
// Double-buffered LDS tiles filled by async global->LDS copies.
#define BMT 128
#define BNT 128
#define BKT 32
__global__ __launch_bounds__(256) void gemm_bf16_kernel(const __bf16* __restrict__ A,
                                                        const __bf16* __restrict__ W,
                                                        const float* __restrict__ bias,
                                                        float* __restrict__ C,
                                                        int M, int N, int K, int has_bias) {
  __shared__ __bf16 As[2][BMT][BKT];
  __shared__ __bf16 Bs[2][BNT][BKT];

  int tid  = threadIdx.x;
  int lane = tid & 31;
  int w    = tid >> 5;          // 0..7
  int wm   = w >> 2;            // 0..1 -> 64-row stripe
  int wn   = w & 3;             // 0..3 -> 32-col stripe
  int m0 = blockIdx.y * BMT;
  int n0 = blockIdx.x * BNT;
  int wm0 = wm * 64;
  int wn0 = wn * 32;

  int row16 = lane & 15;
  int koff  = (lane >> 4) * 8;  // per-lane K sub-offset (ISA 16-bit A layout)

  v8f acc[4][2];
#pragma unroll
  for (int i = 0; i < 4; i++)
#pragma unroll
    for (int j = 0; j < 2; j++)
#pragma unroll
      for (int r = 0; r < 8; r++) acc[i][j][r] = 0.0f;

  // cooperative tile fill: thread -> (row = tid>>1, 16-elem half = tid&1)
  int ldr = tid >> 1;
  int ldh = (tid & 1) * 16;
  // clamp OOB rows/cols into range: clamped data only feeds outputs that the
  // epilogue guards away, and it keeps per-wave ASYNCcnt issue counts uniform.
  int gmc = (m0 + ldr < M) ? (m0 + ldr) : (M - 1);
  int gnc = (n0 + ldr < N) ? (n0 + ldr) : (N - 1);
  const __bf16* gA = A + (size_t)gmc * K + ldh;
  const __bf16* gB = W + (size_t)gnc * K + ldh;

  // async copy of one 32-byte row-segment of each tile (4 instrs/thread/stage)
  auto issue_stage = [&](int buf, int k0) {
    unsigned la = (unsigned)(size_t)(const void*)&As[buf][ldr][ldh];
    unsigned lb = (unsigned)(size_t)(const void*)&Bs[buf][ldr][ldh];
    const void* pa = (const void*)(gA + k0);
    const void* pb = (const void*)(gB + k0);
    asm volatile(
        "global_load_async_to_lds_b128 %0, %1, off\n\t"
        "global_load_async_to_lds_b128 %0, %1, off offset:16\n\t"
        "global_load_async_to_lds_b128 %2, %3, off\n\t"
        "global_load_async_to_lds_b128 %2, %3, off offset:16"
        :: "v"(la), "v"(pa), "v"(lb), "v"(pb)
        : "memory");
  };

  int nsteps = K / BKT;
  issue_stage(0, 0);

  for (int s = 0; s < nsteps; s++) {
    int cur = s & 1;
    if (s + 1 < nsteps) {
      issue_stage(cur ^ 1, (s + 1) * BKT);
      // previous stage's 4 async copies done; next stage's 4 still in flight
      asm volatile("s_wait_asynccnt 0x4" ::: "memory");
    } else {
      asm volatile("s_wait_asynccnt 0x0" ::: "memory");
    }
    __syncthreads();   // all waves' copies for `cur` have landed in LDS

    // ---- fragments ----
    v16bf afrag[4], bfrag[2];
#pragma unroll
    for (int i = 0; i < 4; i++) {
      int r = wm0 + i * 16 + row16;
      v8bf lo = *(const v8bf*)&As[cur][r][koff];
      v8bf hi = *(const v8bf*)&As[cur][r][koff + 16];
#pragma unroll
      for (int e = 0; e < 8; e++) { afrag[i][e] = lo[e]; afrag[i][e + 8] = hi[e]; }
    }
#pragma unroll
    for (int j = 0; j < 2; j++) {
      int c = wn0 + j * 16 + row16;
      v8bf lo = *(const v8bf*)&Bs[cur][c][koff];
      v8bf hi = *(const v8bf*)&Bs[cur][c][koff + 16];
#pragma unroll
      for (int e = 0; e < 8; e++) { bfrag[j][e] = lo[e]; bfrag[j][e + 8] = hi[e]; }
    }

    // ---- 8 WMMAs per wave per k-step ----
#pragma unroll
    for (int i = 0; i < 4; i++)
#pragma unroll
      for (int j = 0; j < 2; j++)
        acc[i][j] = __builtin_amdgcn_wmma_f32_16x16x32_bf16(
            false, afrag[i], false, bfrag[j], (short)0, acc[i][j], false, false);

    __syncthreads();   // done reading `cur` before it is refilled at s+2
  }

  // ---- epilogue: C/D layout -> M = r + 8*(lane>>4), N = lane&15 ----
#pragma unroll
  for (int i = 0; i < 4; i++)
#pragma unroll
    for (int j = 0; j < 2; j++)
#pragma unroll
      for (int r = 0; r < 8; r++) {
        int gm = m0 + wm0 + i * 16 + r + 8 * (lane >> 4);
        int gn = n0 + wn0 + j * 16 + (lane & 15);
        if (gm < M && gn < N) {
          float val = acc[i][j][r];
          if (has_bias) val += bias[gn];
          C[(size_t)gm * N + gn] = val;
        }
      }
}

// ------------------------------------------------------------- project_ball per head chunk (64)
__global__ __launch_bounds__(256) void proj_heads_kernel(float* __restrict__ q, int nchunks) {
  int chunk = blockIdx.x * 8 + (threadIdx.x >> 5);
  int lane  = threadIdx.x & 31;
  if (chunk >= nchunks) return;
  float* p = q + (size_t)chunk * 64;
  float a = p[lane], b = p[lane + 32];
  float s = wred_sum(a * a + b * b);
  float nrm = sqrtf(s);
  if (nrm >= MAXN) {
    float sc = MAXN / fmaxf(nrm, EPSV);
    p[lane] = a * sc;
    p[lane + 32] = b * sc;
  }
}

// ------------------------------------------------------------- attention (f32 VALU; ~1% of FLOPs)
__global__ __launch_bounds__(256) void attn_kernel(const float* __restrict__ qf,
                                                   const float* __restrict__ kf,
                                                   const float* __restrict__ vf,
                                                   __bf16* __restrict__ obf) {
  __shared__ float ks[Lc][65];
  __shared__ float vs[Lc][65];
  __shared__ float k2s[Lc];
  __shared__ float qs[8][64];
  __shared__ float wsm[8][Lc];

  int hh  = blockIdx.x;
  int bb  = blockIdx.y;
  int tid = threadIdx.x;
  int lane = tid & 31, w = tid >> 5;
  int bl0 = bb * Lc;
  int hc  = hh * 64;

  for (int idx = tid; idx < Lc * 64; idx += 256) {
    int m = idx >> 6, d = idx & 63;
    ks[m][d] = kf[(size_t)(bl0 + m) * Dc + hc + d];
    vs[m][d] = vf[(size_t)(bl0 + m) * Dc + hc + d];
  }
  __syncthreads();
  if (tid < Lc) {
    float s = 0.f;
    for (int d = 0; d < 64; d++) s += ks[tid][d] * ks[tid][d];
    k2s[tid] = s;
  }
  __syncthreads();

  for (int l = w; l < Lc; l += 8) {
    float q0 = qf[(size_t)(bl0 + l) * Dc + hc + lane];
    float q1 = qf[(size_t)(bl0 + l) * Dc + hc + 32 + lane];
    qs[w][lane] = q0;
    qs[w][lane + 32] = q1;
    float q2 = wred_sum(q0 * q0 + q1 * q1);

    float sc[4], ev[4];
#pragma unroll
    for (int t = 0; t < 4; t++) {
      int m = lane + t * 32;
      if (m < Lc) {
        float dot = 0.f;
        for (int d = 0; d < 64; d++) dot += qs[w][d] * ks[m][d];
        float diff = fmaxf(q2 + k2s[m] - 2.0f * dot, 0.0f);
        float conf = fmaxf(1.0f - q2, EPSV) * fmaxf(1.0f - k2s[m], EPSV);
        sc[t] = -4.0f * diff / conf;   // -(2*diff/conf)/TEMP, TEMP=0.5
      } else {
        sc[t] = -3.0e38f;
      }
    }
    float mx = fmaxf(fmaxf(sc[0], sc[1]), fmaxf(sc[2], sc[3]));
    mx = wred_max(mx);
    float lsum = 0.f;
#pragma unroll
    for (int t = 0; t < 4; t++) { ev[t] = __expf(sc[t] - mx); lsum += ev[t]; }
    float ssum = wred_sum(lsum);
#pragma unroll
    for (int t = 0; t < 4; t++) {
      int m = lane + t * 32;
      if (m < Lc) wsm[w][m] = ev[t] / ssum;
    }
    float o0 = 0.f, o1 = 0.f;
    for (int m = 0; m < Lc; m++) {
      float wt = wsm[w][m];
      o0 += wt * vs[m][lane];
      o1 += wt * vs[m][lane + 32];
    }
    obf[(size_t)(bl0 + l) * Dc + hc + lane]      = (__bf16)o0;
    obf[(size_t)(bl0 + l) * Dc + hc + 32 + lane] = (__bf16)o1;
  }
}

// ------------------------------------------------------------- residual + layernorm (+bf16)
__global__ __launch_bounds__(256) void add_ln_kernel(const float* __restrict__ hin,
                                                     const float* __restrict__ a,
                                                     const float* __restrict__ g,
                                                     const float* __restrict__ be,
                                                     float* __restrict__ hout,
                                                     __bf16* __restrict__ hbf) {
  __shared__ float red[32];
  int row = blockIdx.x, tid = threadIdx.x;
  const float* hr = hin + (size_t)row * Dc;
  const float* ar = a + (size_t)row * Dc;
  float xv[4]; float s = 0.f, s2 = 0.f;
#pragma unroll
  for (int j = 0; j < 4; j++) {
    int c = tid + j * 256;
    float t = hr[c] + ar[c];
    xv[j] = t; s += t; s2 += t * t;
  }
  s = block_sum_256(s, red);
  s2 = block_sum_256(s2, red);
  float mean = s / (float)Dc;
  float var  = s2 / (float)Dc - mean * mean;
  float inv  = rsqrtf(var + 1e-5f);
#pragma unroll
  for (int j = 0; j < 4; j++) {
    int c = tid + j * 256;
    float y = (xv[j] - mean) * inv * g[c] + be[c];
    hout[(size_t)row * Dc + c] = y;
    hbf[(size_t)row * Dc + c]  = (__bf16)y;
  }
}

// residual + layernorm + project_ball (+bf16)
__global__ __launch_bounds__(256) void add_ln_proj_kernel(const float* __restrict__ hin,
                                                          const float* __restrict__ a,
                                                          const float* __restrict__ g,
                                                          const float* __restrict__ be,
                                                          float* __restrict__ hout,
                                                          __bf16* __restrict__ hbf) {
  __shared__ float red[32];
  int row = blockIdx.x, tid = threadIdx.x;
  const float* hr = hin + (size_t)row * Dc;
  const float* ar = a + (size_t)row * Dc;
  float xv[4]; float s = 0.f, s2 = 0.f;
#pragma unroll
  for (int j = 0; j < 4; j++) {
    int c = tid + j * 256;
    float t = hr[c] + ar[c];
    xv[j] = t; s += t; s2 += t * t;
  }
  s = block_sum_256(s, red);
  s2 = block_sum_256(s2, red);
  float mean = s / (float)Dc;
  float var  = s2 / (float)Dc - mean * mean;
  float inv  = rsqrtf(var + 1e-5f);
  float yv[4]; float n2 = 0.f;
#pragma unroll
  for (int j = 0; j < 4; j++) {
    int c = tid + j * 256;
    yv[j] = (xv[j] - mean) * inv * g[c] + be[c];
    n2 += yv[j] * yv[j];
  }
  n2 = block_sum_256(n2, red);
  float nrm = sqrtf(n2);
  float sc  = (nrm < MAXN) ? 1.0f : MAXN / fmaxf(nrm, EPSV);
#pragma unroll
  for (int j = 0; j < 4; j++) {
    int c = tid + j * 256;
    float y = yv[j] * sc;
    hout[(size_t)row * Dc + c] = y;
    hbf[(size_t)row * Dc + c]  = (__bf16)y;
  }
}

// ------------------------------------------------------------- Einstein midpoint pool (per batch b)
__global__ __launch_bounds__(256) void pool_kernel(const float* __restrict__ h,
                                                   float* __restrict__ pooled,
                                                   __bf16* __restrict__ pooled_bf,
                                                   float* __restrict__ dx) {
  __shared__ float gam[Lc];
  __shared__ float red[32];
  int b = blockIdx.x, tid = threadIdx.x;
  if (tid < Lc) {
    const float* row = h + (size_t)(b * Lc + tid) * Dc;
    float s = 0.f;
    for (int c = 0; c < Dc; c++) s += row[c] * row[c];
    gam[tid] = 1.0f / fmaxf(1.0f - s, EPSV);
  }
  __syncthreads();
  float gs = (tid < Lc) ? gam[tid] : 0.f;
  gs = block_sum_256(gs, red);
  float mid[4];
#pragma unroll
  for (int j = 0; j < 4; j++) {
    int c = tid + j * 256;
    float s = 0.f;
    for (int l = 0; l < Lc; l++) s += gam[l] * h[(size_t)(b * Lc + l) * Dc + c];
    mid[j] = s / fmaxf(gs, EPSV);
  }
  float n2 = 0.f;
#pragma unroll
  for (int j = 0; j < 4; j++) n2 += mid[j] * mid[j];
  n2 = block_sum_256(n2, red);
  float mn = sqrtf(n2);
  float sc = (mn < MAXN) ? 1.0f : MAXN / fmaxf(mn, EPSV);
#pragma unroll
  for (int j = 0; j < 4; j++) {
    int c = tid + j * 256;
    float v = mid[j] * sc;
    pooled[(size_t)b * Dc + c]    = v;
    pooled_bf[(size_t)b * Dc + c] = (__bf16)v;
  }
  if (tid == 0) dx[b] = n2 * sc * sc;
}

// ------------------------------------------------------------- prototype projection
__global__ __launch_bounds__(256) void proto_prep_kernel(const float* __restrict__ protos,
                                                         __bf16* __restrict__ pbf,
                                                         float* __restrict__ dy) {
  __shared__ float red[32];
  int c = blockIdx.x, tid = threadIdx.x;
  const float* pr = protos + (size_t)c * Dc;
  float e[4]; float s = 0.f;
#pragma unroll
  for (int j = 0; j < 4; j++) { int k = tid + j * 256; e[j] = pr[k]; s += e[j] * e[j]; }
  s = block_sum_256(s, red);
  float nrm = sqrtf(s);
  float sc = (nrm < MAXN) ? 1.0f : MAXN / fmaxf(nrm, EPSV);
#pragma unroll
  for (int j = 0; j < 4; j++) {
    int k = tid + j * 256;
    pbf[(size_t)c * Dc + k] = (__bf16)(e[j] * sc);
  }
  if (tid == 0) dy[c] = s * sc * sc;
}

// ------------------------------------------------------------- Poincare distance + output
__global__ __launch_bounds__(256) void final_kernel(const float* __restrict__ S,
                                                    const float* __restrict__ dx,
                                                    const float* __restrict__ dy,
                                                    const float* __restrict__ temp,
                                                    float* __restrict__ out, int n) {
  int i = blockIdx.x * 256 + threadIdx.x;
  if (i >= n) return;
  int b = i / Cc, c = i % Cc;
  float dxb = dx[b], dyc = dy[c];
  float diff  = fmaxf(dxb + dyc - 2.0f * S[i], 0.0f);
  float delta = 2.0f * diff / (fmaxf(1.0f - dxb, EPSV) * fmaxf(1.0f - dyc, EPSV));
  float dist  = logf(1.0f + delta + sqrtf(fmaxf(delta * (delta + 2.0f), EPSV)));
  out[i] = -dist / fmaxf(fabsf(temp[0]), 0.1f);
}

// ===========================================================================
extern "C" void kernel_launch(void* const* d_in, const int* in_sizes, int n_in,
                              void* d_out, int out_size, void* d_ws, size_t ws_size,
                              hipStream_t stream) {
  (void)in_sizes; (void)n_in; (void)out_size; (void)ws_size;
  const int*   x    = (const int*)d_in[0];
  const float* emb  = (const float*)d_in[1];
  const float* pos  = (const float*)d_in[2];
  const float* Wq   = (const float*)d_in[3];
  const float* bq   = (const float*)d_in[4];
  const float* Wk   = (const float*)d_in[5];
  const float* bk   = (const float*)d_in[6];
  const float* Wv   = (const float*)d_in[7];
  const float* bv   = (const float*)d_in[8];
  const float* Wo   = (const float*)d_in[9];
  const float* bo   = (const float*)d_in[10];
  const float* W1   = (const float*)d_in[11];
  const float* b1f  = (const float*)d_in[12];
  const float* W2   = (const float*)d_in[13];
  const float* b2f  = (const float*)d_in[14];
  const float* g1   = (const float*)d_in[15];
  const float* be1  = (const float*)d_in[16];
  const float* g2   = (const float*)d_in[17];
  const float* be2  = (const float*)d_in[18];
  const float* prot = (const float*)d_in[19];
  const float* temp = (const float*)d_in[20];

  char* base = (char*)d_ws;
  size_t off = 0;
  auto carve = [&](size_t bytes) -> void* {
    void* p = base + off;
    off = (off + bytes + 255) & ~(size_t)255;
    return p;
  };

  float*  h    = (float*)carve((size_t)Mc * Dc * 4);
  __bf16* hbf  = (__bf16*)carve((size_t)Mc * Dc * 2);
  float*  qf   = (float*)carve((size_t)Mc * Dc * 4);
  float*  kf   = (float*)carve((size_t)Mc * Dc * 4);
  float*  vf   = (float*)carve((size_t)Mc * Dc * 4);
  float*  af   = (float*)carve((size_t)Mc * Dc * 4);
  __bf16* obf  = (__bf16*)carve((size_t)Mc * Dc * 2);
  float*  ffn1 = (float*)carve((size_t)Mc * DFc * 4);
  __bf16* gbf  = (__bf16*)carve((size_t)Mc * DFc * 2);
  __bf16* wq_bf = (__bf16*)carve((size_t)NLc * Dc * Dc * 2);
  __bf16* wk_bf = (__bf16*)carve((size_t)NLc * Dc * Dc * 2);
  __bf16* wv_bf = (__bf16*)carve((size_t)NLc * Dc * Dc * 2);
  __bf16* wo_bf = (__bf16*)carve((size_t)NLc * Dc * Dc * 2);
  __bf16* w1_bf = (__bf16*)carve((size_t)NLc * DFc * Dc * 2);
  __bf16* w2_bf = (__bf16*)carve((size_t)NLc * DFc * Dc * 2);
  __bf16* pbf   = (__bf16*)carve((size_t)Cc * Dc * 2);
  float*  pooled    = (float*)carve((size_t)Bc * Dc * 4);
  __bf16* pooled_bf = (__bf16*)carve((size_t)Bc * Dc * 2);
  float*  dx = (float*)carve((size_t)Bc * 4);
  float*  dy = (float*)carve((size_t)Cc * 4);
  float*  S  = (float*)carve((size_t)Bc * Cc * 4);

  auto cvt = [&](const float* in, __bf16* o, size_t n) {
    cvt_bf16_kernel<<<dim3((unsigned)((n + 255) / 256)), dim3(256), 0, stream>>>(in, o, (int)n);
  };
  auto gemm = [&](const __bf16* A, const __bf16* Wm, const float* bias, float* Cout,
                  int M_, int N_, int K_) {
    dim3 g((N_ + BNT - 1) / BNT, (M_ + BMT - 1) / BMT);
    gemm_bf16_kernel<<<g, dim3(256), 0, stream>>>(A, Wm, bias, Cout, M_, N_, K_,
                                                  bias != nullptr ? 1 : 0);
  };

  // weight conversion (once per launch)
  cvt(Wq, wq_bf, (size_t)NLc * Dc * Dc);
  cvt(Wk, wk_bf, (size_t)NLc * Dc * Dc);
  cvt(Wv, wv_bf, (size_t)NLc * Dc * Dc);
  cvt(Wo, wo_bf, (size_t)NLc * Dc * Dc);
  cvt(W1, w1_bf, (size_t)NLc * DFc * Dc);
  cvt(W2, w2_bf, (size_t)NLc * DFc * Dc);
  proto_prep_kernel<<<dim3(Cc), dim3(256), 0, stream>>>(prot, pbf, dy);

  // embedding + exp map
  embed_kernel<<<dim3(Mc), dim3(256), 0, stream>>>(x, emb, pos, h, hbf);

  for (int i = 0; i < NLc; i++) {
    const __bf16* wqi = wq_bf + (size_t)i * Dc * Dc;
    const __bf16* wki = wk_bf + (size_t)i * Dc * Dc;
    const __bf16* wvi = wv_bf + (size_t)i * Dc * Dc;
    const __bf16* woi = wo_bf + (size_t)i * Dc * Dc;
    const __bf16* w1i = w1_bf + (size_t)i * DFc * Dc;
    const __bf16* w2i = w2_bf + (size_t)i * DFc * Dc;

    gemm(hbf, wqi, bq + (size_t)i * Dc, qf, Mc, Dc, Dc);
    gemm(hbf, wki, bk + (size_t)i * Dc, kf, Mc, Dc, Dc);
    gemm(hbf, wvi, bv + (size_t)i * Dc, vf, Mc, Dc, Dc);

    int nchunks = Mc * Hc;
    proj_heads_kernel<<<dim3((nchunks + 7) / 8), dim3(256), 0, stream>>>(qf, nchunks);
    proj_heads_kernel<<<dim3((nchunks + 7) / 8), dim3(256), 0, stream>>>(kf, nchunks);

    attn_kernel<<<dim3(Hc, Bc), dim3(256), 0, stream>>>(qf, kf, vf, obf);

    gemm(obf, woi, bo + (size_t)i * Dc, af, Mc, Dc, Dc);
    add_ln_kernel<<<dim3(Mc), dim3(256), 0, stream>>>(h, af, g1 + (size_t)i * Dc,
                                                      be1 + (size_t)i * Dc, h, hbf);

    gemm(hbf, w1i, b1f + (size_t)i * DFc, ffn1, Mc, DFc, Dc);
    {
      size_t n = (size_t)Mc * DFc;
      gelu_cvt_kernel<<<dim3((unsigned)((n + 255) / 256)), dim3(256), 0, stream>>>(ffn1, gbf, (int)n);
    }
    gemm(gbf, w2i, b2f + (size_t)i * Dc, af, Mc, Dc, DFc);
    add_ln_proj_kernel<<<dim3(Mc), dim3(256), 0, stream>>>(h, af, g2 + (size_t)i * Dc,
                                                           be2 + (size_t)i * Dc, h, hbf);
  }

  // pooling + prototype distances
  pool_kernel<<<dim3(Bc), dim3(256), 0, stream>>>(h, pooled, pooled_bf, dx);
  gemm(pooled_bf, pbf, nullptr, S, Bc, Cc, Dc);
  {
    int n = Bc * Cc;
    final_kernel<<<dim3((n + 255) / 256), dim3(256), 0, stream>>>(S, dx, dy, temp,
                                                                  (float*)d_out, n);
  }
}
